// GloRe_Unit_79242146611997
// MI455X (gfx1250) — matrix operand order repaired and checked by reading, not compiled, not run on hardware
//
#include <hip/hip_runtime.h>
#include <cstdint>

// ---------------------------------------------------------------------------
// GloRe unit on gfx1250 (MI455X): bf16 WMMA GEMMs + f32 softmax.
// v2: 32x64 wave tiles (8 wmma / K-step), double-buffered LDS (1 barrier /
// K-step), and GLOBAL_LOAD_ASYNC_TO_LDS_B128 (ASYNCcnt path) for all pure
// bf16 tile copies so the TDM-style DMA overlaps WMMA compute.
// ---------------------------------------------------------------------------

typedef __bf16 bf16_t;
typedef __attribute__((ext_vector_type(16))) __bf16 bf16x16;
typedef __attribute__((ext_vector_type(4)))  __bf16 bf16x4;
typedef __attribute__((ext_vector_type(8)))  float  f32x8;

#define BATCH 16
#define C_IN  512
#define MM    256      // M_MID
#define HW    4096

#define LDS_SA 40      // A tile LDS row stride (elems); 64 rows x 32 cols
#define LDS_SB 136     // B tile LDS row stride (elems); 32 rows x 128 cols
#define A_BUF  (64 * LDS_SA)
#define B_BUF  (32 * LDS_SB)

union Frag16 {
  bf16x16 v;
  uint4   q[2];
  bf16_t  e[16];
};

// Async global->LDS 16B copy per lane (tracked by ASYNCcnt, no VGPR data path).
__device__ __forceinline__ void async_copy16(uint32_t lds_byte_off, const void* gp) {
  asm volatile("global_load_async_to_lds_b128 %0, %1, off"
               :: "v"(lds_byte_off), "v"((uint64_t)(uintptr_t)gp)
               : "memory");
}
__device__ __forceinline__ void wait_async() {
  asm volatile("s_wait_asynccnt 0" ::: "memory");
}
__device__ __forceinline__ uint32_t lds_off(const void* p) {
  return (uint32_t)(uintptr_t)p;   // LDS aperture lives in addr[63:32]
}

// A fragment (16x32 bf16, MxK). lane%16 = M row; lanes 0-15: K={0..7,16..23},
// lanes 16-31: K={8..15,24..31}.
__device__ __forceinline__ bf16x16 lds_frag_a(const bf16_t* lsA, int m_base, int lane) {
  const int row  = m_base + (lane & 15);
  const int half = (lane >> 4) << 3;
  const bf16_t* p = lsA + row * LDS_SA;
  Frag16 f;
  f.q[0] = *(const uint4*)(p + half);
  f.q[1] = *(const uint4*)(p + 16 + half);
  return f.v;
}

// B fragment (32x16 bf16, KxN). lane%16 = N column; lanes 0-15: K={0..7,16..23},
// lanes 16-31: K={8..15,24..31}.
__device__ __forceinline__ bf16x16 lds_frag_b(const bf16_t* lsB, int n_base, int lane) {
  const int half = (lane >> 4) << 3;
  const bf16_t* p = lsB + n_base + (lane & 15);
  Frag16 f;
  #pragma unroll
  for (int s = 0; s < 8; ++s) {
    f.e[s]     = p[(half + s) * LDS_SB];
    f.e[8 + s] = p[(16 + half + s) * LDS_SB];
  }
  return f.v;
}

// ---------------------------------------------------------------------------
// Unified tiled WMMA GEMM. Block = 128 threads (4 wave32) -> 64x128 tile;
// each wave owns 32x64 (2x4 WMMA tiles); K-step 32, double-buffered LDS.
//   MODE 0: S/P = bf16(W[256x512] @ x_f32[512x4096] + bias)   (z = batch*2+which)
//   MODE 1: scores_f32 = (S[256x4096] @ P^T) / 64             (B transposed in LDS)
//   MODE 2: R = bf16(attn[256x256] @ P[256x4096])
//   MODE 3: out_f32 = x + Wext[512x256] @ R[256x4096] + bias
// ---------------------------------------------------------------------------
template <int MODE>
__global__ __launch_bounds__(128) void gemm_wmma(
    const bf16_t* __restrict__ A0,
    const bf16_t* __restrict__ A1,
    const float*  __restrict__ Bsrc_f,
    const bf16_t* __restrict__ Bsrc_b,
    const float*  __restrict__ bias0,
    const float*  __restrict__ bias1,
    bf16_t*       __restrict__ Out_b0,
    bf16_t*       __restrict__ Out_b1,
    float*        __restrict__ Out_f,
    const float*  __restrict__ Xres)
{
  constexpr int Kdim      = (MODE == 0) ? C_IN : (MODE == 1) ? HW : MM;
  constexpr int OutStride = (MODE == 1) ? MM : HW;
  constexpr int NIT       = Kdim / 32;

  __shared__ bf16_t lsA[2 * A_BUF];
  __shared__ bf16_t lsB[2 * B_BUF];

  const int t    = threadIdx.x;
  const int lane = t & 31;
  const int wave = t >> 5;
  const int wr0  = (wave >> 1) << 5;   // 0 / 32  (M within block tile)
  const int wc0  = (wave & 1)  << 6;   // 0 / 64  (N within block tile)

  const int n0 = blockIdx.x * 128;
  const int m0 = blockIdx.y * 64;
  const int z  = blockIdx.z;

  const int batch = (MODE == 0) ? (z >> 1) : z;
  const int which = (MODE == 0) ? (z & 1)  : 0;

  const bf16_t* A    = nullptr;
  const float*  Bf   = nullptr;
  const bf16_t* Bb   = nullptr;
  const float*  bias = nullptr;
  bf16_t*       Outb = nullptr;
  float*        Outf = nullptr;
  const float*  Xr   = nullptr;

  if constexpr (MODE == 0) {
    A    = which ? A1 : A0;
    Bf   = Bsrc_f + (size_t)batch * C_IN * HW;
    bias = which ? bias1 : bias0;
    Outb = (which ? Out_b1 : Out_b0) + (size_t)batch * MM * HW;
  } else if constexpr (MODE == 1) {
    A    = A0 + (size_t)batch * MM * HW;
    Bb   = Bsrc_b + (size_t)batch * MM * HW;
    Outf = Out_f + (size_t)batch * MM * MM;
  } else if constexpr (MODE == 2) {
    A    = A0 + (size_t)batch * MM * MM;
    Bb   = Bsrc_b + (size_t)batch * MM * HW;
    Outb = Out_b0 + (size_t)batch * MM * HW;
  } else {
    A    = A0;
    Bb   = Bsrc_b + (size_t)batch * MM * HW;
    bias = bias0;
    Outf = Out_f + (size_t)batch * C_IN * HW;
    Xr   = Xres  + (size_t)batch * C_IN * HW;
  }

  // ---- staging lambdas -----------------------------------------------------
  // A tile [m0,m0+64) x [kk,kk+32): pure bf16 copy -> async DMA to LDS.
  auto stageA = [&](int kk, int buf) {
    const int row = t >> 1;                    // 0..63
    const int col = (t & 1) << 4;              // 0 / 16
    const bf16_t* gp = A + (size_t)(m0 + row) * Kdim + kk + col;
    const uint32_t lo = lds_off(lsA + buf * A_BUF + row * LDS_SA + col);
    async_copy16(lo,      gp);
    async_copy16(lo + 16, gp + 8);
  };
  // B tile [kk,kk+32) x [n0,n0+128) direct bf16 copy (MODE 2/3) -> async DMA.
  auto stageB_async = [&](int kk, int buf) {
    const int row = t >> 2;                    // 0..31
    const int col = (t & 3) << 5;              // 0,32,64,96
    const bf16_t* gp = Bb + (size_t)(kk + row) * HW + n0 + col;
    const uint32_t lo = lds_off(lsB + buf * B_BUF + row * LDS_SB + col);
    #pragma unroll
    for (int i = 0; i < 4; ++i) async_copy16(lo + 16 * i, gp + 8 * i);
  };

  float4 fB[8];   // MODE0 staging (f32 -> bf16 conversion path)
  uint4  uB[4];   // MODE1 staging (transpose path)

  auto issueB = [&](int kk) {
    if constexpr (MODE == 0) {
      const int row = t >> 2;                  // 0..31
      const int col = (t & 3) << 5;            // 0,32,64,96
      const float* gp = Bf + (size_t)(kk + row) * HW + n0 + col;
      #pragma unroll
      for (int i = 0; i < 8; ++i) fB[i] = *(const float4*)(gp + 4 * i);
    } else if constexpr (MODE == 1) {
      const int nidx = t;                      // 0..127 (row of P)
      const bf16_t* gp = Bb + (size_t)(n0 + nidx) * HW + kk;
      #pragma unroll
      for (int i = 0; i < 4; ++i) uB[i] = *(const uint4*)(gp + 8 * i);
    }
  };
  auto commitB = [&](int buf) {
    if constexpr (MODE == 0) {
      const int row = t >> 2;
      const int col = (t & 3) << 5;
      bf16_t* pB = lsB + buf * B_BUF + row * LDS_SB + col;
      #pragma unroll
      for (int i = 0; i < 8; ++i) {
        bf16x4 pk = { (bf16_t)fB[i].x, (bf16_t)fB[i].y,
                      (bf16_t)fB[i].z, (bf16_t)fB[i].w };
        *(bf16x4*)(pB + 4 * i) = pk;
      }
    } else if constexpr (MODE == 1) {
      const int nidx = t;
      union { uint4 q[4]; bf16_t e[32]; } u;
      #pragma unroll
      for (int i = 0; i < 4; ++i) u.q[i] = uB[i];
      bf16_t* pB = lsB + buf * B_BUF + nidx;
      #pragma unroll
      for (int e = 0; e < 32; ++e) pB[e * LDS_SB] = u.e[e];
    }
  };

  // ---- prologue: stage tile 0 into buffer 0 --------------------------------
  stageA(0, 0);
  if constexpr (MODE >= 2) stageB_async(0, 0);
  else { issueB(0); commitB(0); }
  wait_async();
  __syncthreads();

  f32x8 acc[2][4] = {};

  for (int it = 0; it < NIT; ++it) {
    const int  cur  = it & 1;
    const int  nxt  = cur ^ 1;
    const bool more = (it + 1 < NIT);

    if (more) {                                 // issue next tile (overlaps wmma)
      stageA((it + 1) * 32, nxt);
      if constexpr (MODE >= 2) stageB_async((it + 1) * 32, nxt);
      else issueB((it + 1) * 32);
    }

    // compute from current buffer
    const bf16_t* cA = lsA + cur * A_BUF;
    const bf16_t* cB = lsB + cur * B_BUF;
    bf16x16 af[2], bfr[4];
    af[0] = lds_frag_a(cA, wr0,      lane);
    af[1] = lds_frag_a(cA, wr0 + 16, lane);
    #pragma unroll
    for (int j = 0; j < 4; ++j) bfr[j] = lds_frag_b(cB, wc0 + 16 * j, lane);

    #pragma unroll
    for (int ti = 0; ti < 2; ++ti)
      #pragma unroll
      for (int tj = 0; tj < 4; ++tj)
        acc[ti][tj] = __builtin_amdgcn_wmma_f32_16x16x32_bf16(
            false, af[ti], false, bfr[tj], (short)0, acc[ti][tj], false, false);

    if (more) {
      if constexpr (MODE < 2) commitB(nxt);     // cvt/transpose into next buffer
      wait_async();
      __syncthreads();
    }
  }

  // ---- epilogue: C/D layout -> M = r + 8*(lane>=16), N = lane&15
  const int rAdd = (lane >> 4) << 3;
  const int ln   = lane & 15;
  #pragma unroll
  for (int ti = 0; ti < 2; ++ti) {
    #pragma unroll
    for (int tj = 0; tj < 4; ++tj) {
      #pragma unroll
      for (int r = 0; r < 8; ++r) {
        const int gm = m0 + wr0 + ti * 16 + rAdd + r;
        const int gn = n0 + wc0 + tj * 16 + ln;
        const float v = acc[ti][tj][r];
        if constexpr (MODE == 0) {
          Outb[(size_t)gm * OutStride + gn] = (bf16_t)(v + bias[gm]);
        } else if constexpr (MODE == 1) {
          Outf[(size_t)gm * OutStride + gn] = v * 0.015625f;   // 1/sqrt(4096)
        } else if constexpr (MODE == 2) {
          Outb[(size_t)gm * OutStride + gn] = (bf16_t)v;
        } else {
          const size_t idx = (size_t)gm * OutStride + gn;
          Outf[idx] = Xr[idx] + v + bias[gm];
        }
      }
    }
  }
}

// ---------------------------------------------------------------------------
// Row softmax over scores [16*256 rows][256]: one wave32 per row.
// ---------------------------------------------------------------------------
__global__ __launch_bounds__(256) void softmax_rows(const float* __restrict__ scores,
                                                    bf16_t* __restrict__ attnb)
{
  const int wave = threadIdx.x >> 5;
  const int lane = threadIdx.x & 31;
  const int row  = blockIdx.x * 8 + wave;
  const float* p = scores + (size_t)row * MM;

  float v[8];
  float mx = -3.4e38f;
  #pragma unroll
  for (int i = 0; i < 8; ++i) { v[i] = p[lane + i * 32]; mx = fmaxf(mx, v[i]); }
  #pragma unroll
  for (int off = 16; off > 0; off >>= 1) mx = fmaxf(mx, __shfl_xor(mx, off, 32));

  float sum = 0.0f;
  #pragma unroll
  for (int i = 0; i < 8; ++i) { v[i] = __expf(v[i] - mx); sum += v[i]; }
  #pragma unroll
  for (int off = 16; off > 0; off >>= 1) sum += __shfl_xor(sum, off, 32);

  const float inv = 1.0f / sum;
  bf16_t* o = attnb + (size_t)row * MM;
  #pragma unroll
  for (int i = 0; i < 8; ++i) o[lane + i * 32] = (bf16_t)(v[i] * inv);
}

// ---------------------------------------------------------------------------
// Convert the three weight matrices f32 -> bf16 (contiguous in workspace).
// ---------------------------------------------------------------------------
__global__ __launch_bounds__(256) void cvt_weights(const float* __restrict__ Ws,
                                                   const float* __restrict__ Wp,
                                                   const float* __restrict__ We,
                                                   bf16_t* __restrict__ out)
{
  const int i = blockIdx.x * 256 + threadIdx.x;      // 0 .. 393215
  const float* src; int j = i;
  if (i < 131072)       { src = Ws; }
  else if (i < 262144)  { src = Wp; j = i - 131072; }
  else                  { src = We; j = i - 262144; }
  out[i] = (bf16_t)src[j];
}

// ---------------------------------------------------------------------------
// Host launcher. Workspace layout (bytes), total ~74 MB (R aliases S):
//   0        : Wsb  bf16  262144
//   262144   : Wpb  bf16  262144
//   524288   : Web  bf16  262144
//   786432   : S    bf16  33554432   (reused as R after scores are computed)
//   34340864 : P    bf16  33554432
//   67895296 : scores f32  4194304
//   72089600 : attn bf16   2097152
// ---------------------------------------------------------------------------
extern "C" void kernel_launch(void* const* d_in, const int* in_sizes, int n_in,
                              void* d_out, int out_size, void* d_ws, size_t ws_size,
                              hipStream_t stream)
{
  const float* x        = (const float*)d_in[0];
  const float* W_state  = (const float*)d_in[1];
  const float* b_state  = (const float*)d_in[2];
  const float* W_proj   = (const float*)d_in[3];
  const float* b_proj   = (const float*)d_in[4];
  const float* W_extend = (const float*)d_in[5];
  const float* b_extend = (const float*)d_in[6];
  float* out = (float*)d_out;

  char* ws = (char*)d_ws;
  bf16_t* Wsb = (bf16_t*)(ws + 0);
  bf16_t* Wpb = (bf16_t*)(ws + 262144);
  bf16_t* Web = (bf16_t*)(ws + 524288);
  bf16_t* S   = (bf16_t*)(ws + 786432);
  bf16_t* R   = S;                                  // alias: S dead after MODE1
  bf16_t* P   = (bf16_t*)(ws + 34340864);
  float*  Sc  = (float*) (ws + 67895296);
  bf16_t* At  = (bf16_t*)(ws + 72089600);

  // 1) weights -> bf16
  cvt_weights<<<1536, 256, 0, stream>>>(W_state, W_proj, W_extend, Wsb);

  // 2) S,P = W @ x + b   (M=256, N=4096, K=512; z = batch*2 + which)
  gemm_wmma<0><<<dim3(32, 4, 32), 128, 0, stream>>>(
      Wsb, Wpb, x, nullptr, b_state, b_proj, S, P, nullptr, nullptr);

  // 3) scores = S @ P^T / 64   (M=256, N=256, K=4096)
  gemm_wmma<1><<<dim3(2, 4, 16), 128, 0, stream>>>(
      S, nullptr, nullptr, P, nullptr, nullptr, nullptr, nullptr, Sc, nullptr);

  // 4) attn = softmax(scores)  -> bf16
  softmax_rows<<<512, 256, 0, stream>>>(Sc, At);

  // 5) R = attn @ P            (M=256, N=4096, K=256)
  gemm_wmma<2><<<dim3(32, 4, 16), 128, 0, stream>>>(
      At, nullptr, nullptr, P, nullptr, nullptr, R, nullptr, nullptr, nullptr);

  // 6) out = x + Wext @ R + b  (M=512, N=4096, K=256)
  gemm_wmma<3><<<dim3(32, 8, 16), 128, 0, stream>>>(
      Web, nullptr, nullptr, R, b_extend, nullptr, nullptr, nullptr, out, x);
}
